// MultiHeadedAttention_1726576856746
// MI455X (gfx1250) — matrix operand order repaired and checked by reading, not compiled, not run on hardware
//
#include <hip/hip_runtime.h>
#include <hip/hip_bf16.h>
#include <stdint.h>

typedef __bf16 bf16_t;
typedef __attribute__((ext_vector_type(16))) __bf16 v16bf;
typedef __attribute__((ext_vector_type(8)))  float  v8f;
typedef __attribute__((ext_vector_type(4)))  unsigned int v4u;
typedef __attribute__((ext_vector_type(8)))  int v8i;
typedef __attribute__((ext_vector_type(4)))  int v4i;

#define NUM_HEADS 16
#define HEAD      64
#define DMODEL    1024
#define MSEQ      2048
#define BATCH     2

#if __has_builtin(__builtin_amdgcn_global_load_async_to_lds_b128)
#define HAVE_ASYNC 1
#else
#define HAVE_ASYNC 0
#endif
#if __has_builtin(__builtin_amdgcn_tensor_load_to_lds)
#define HAVE_TDM 1
#else
#define HAVE_TDM 0
#endif

#define AS1 __attribute__((address_space(1)))
#define AS3 __attribute__((address_space(3)))

#if HAVE_ASYNC
// 16B global -> LDS async copy (tracked by ASYNCcnt).
// Builtin signature (probed): void(v4i __global*, v4i __shared*, Ii offset, Ii cpol)
__device__ __forceinline__ void async_cp16(const float* g, void* l) {
  __builtin_amdgcn_global_load_async_to_lds_b128(
      (AS1 v4i*)(uintptr_t)g, (AS3 v4i*)(uint32_t)(uintptr_t)l, 0, 0);
}
#endif

template <int N> __device__ __forceinline__ void wait_async() {
#if __has_builtin(__builtin_amdgcn_s_wait_asynccnt)
  __builtin_amdgcn_s_wait_asynccnt(N);
#else
  asm volatile("s_wait_asynccnt %0" :: "n"(N) : "memory");
#endif
}

#if HAVE_TDM
// TDM: DMA a 2-D fp32 tile (rows x cols, row stride ld elements) from global into
// LDS whose rows are padded to 36 dwords (pad_interval=32dw -> code 4, amount=4dw -> code 3).
// D# layout per CDNA5 ISA 8.3/8.4. One instruction per wave; tracked by TENSORcnt.
__device__ __forceinline__ void tdm_load_tile(void* lds, const float* g,
                                              unsigned rows, unsigned cols, unsigned ld) {
  uint64_t ga = (uint64_t)(uintptr_t)g;
  v4u g0;
  g0[0] = 1u;                                                  // count=1, user mode
  g0[1] = (unsigned)(uint32_t)(uintptr_t)lds;                  // lds_addr
  g0[2] = (unsigned)ga;                                        // global_addr[31:0]
  g0[3] = (unsigned)((ga >> 32) & 0x1FFFFFFu) | (2u << 30);    // global_addr[56:32] | type=2
  v8i g1;
  g1[0] = (int)((2u << 16) | (1u << 20) | (4u << 22) | (3u << 25)); // 4B elems, pad 4dw/32dw
  g1[1] = (int)((ld & 0xFFFFu) << 16);                         // tensor_dim0[15:0]
  g1[2] = (int)(((ld >> 16) & 0xFFFFu) | ((rows & 0xFFFFu) << 16)); // dim0 hi | tensor_dim1 lo
  g1[3] = (int)(((rows >> 16) & 0xFFFFu) | ((cols & 0xFFFFu) << 16)); // dim1 hi | tile_dim0
  g1[4] = (int)(rows & 0xFFFFu);                               // tile_dim1 (tile_dim2=0)
  g1[5] = (int)ld;                                             // tensor_dim0_stride[31:0]
  g1[6] = 0;                                                   // stride hi | dim1_stride lo
  g1[7] = 0;
  v4i gz = {0, 0, 0, 0};
#if __clang_major__ >= 23
  v8i gz8 = {0, 0, 0, 0, 0, 0, 0, 0};
  __builtin_amdgcn_tensor_load_to_lds(g0, g1, gz, gz, gz8, 0);
#else
  __builtin_amdgcn_tensor_load_to_lds(g0, g1, gz, gz, 0);
#endif
}
#endif

__device__ __forceinline__ bf16_t to_bf16(float x) { return (bf16_t)x; }

// A-fragment (16x32 bf16) from bf16 K-major row: K = {koff..+7, koff+16..+23}
__device__ __forceinline__ v16bf a_frag(const bf16_t* row, int koff) {
  v16bf f;
#pragma unroll
  for (int i = 0; i < 8; ++i) { f[i] = row[koff + i]; f[i + 8] = row[koff + 16 + i]; }
  return f;
}
// B-fragment (32x16 bf16) from bf16 K-major column data (caller pre-offsets kbase)
__device__ __forceinline__ v16bf b_frag(const bf16_t* col) {
  v16bf f;
#pragma unroll
  for (int i = 0; i < 8; ++i) { f[i] = col[i]; f[i + 8] = col[16 + i]; }
  return f;
}
// Same fragments built from fp32 LDS (async/TDM staged raw), converting on read
__device__ __forceinline__ v16bf a_frag_f32(const float* row, int koff) {
  v16bf f;
#pragma unroll
  for (int i = 0; i < 8; ++i) { f[i] = to_bf16(row[koff + i]); f[i + 8] = to_bf16(row[koff + 16 + i]); }
  return f;
}
__device__ __forceinline__ v16bf b_frag_f32(const float* col) {
  v16bf f;
#pragma unroll
  for (int i = 0; i < 8; ++i) { f[i] = to_bf16(col[i]); f[i + 8] = to_bf16(col[16 + i]); }
  return f;
}

__device__ __forceinline__ v8f wmma_bf16(v16bf a, v16bf b, v8f c) {
  return __builtin_amdgcn_wmma_f32_16x16x32_bf16(false, a, false, b, (short)0, c, false, false);
}

// Y = X[R,K] @ W[N,K]^T + bias; 64x64 tile per 128-thread WG, double-buffered LDS,
// A staged via async-to-LDS, B staged via TDM (fallbacks if builtins absent).
// mode 0: Y[r*N+n];  mode 1: scatter into [B,H,M,D] with scale.
__global__ __launch_bounds__(128) void proj_gemm(const float* __restrict__ X,
                                                 const float* __restrict__ W,
                                                 const float* __restrict__ bias,
                                                 float* __restrict__ Y,
                                                 int K, int N, int mode, float scale) {
  __shared__ float Asf[2][64][36];
  __shared__ float Bsf[2][64][36];
  const int t = threadIdx.x;
  const int lane = t & 31, w = t >> 5;
  const int rowBase = blockIdx.x * 64;
  const int colBase = blockIdx.y * 64;
  const int wrow = (w & 1) * 32, wcol = (w >> 1) * 32;
  const int lrow = lane & 15;
  const int ahalf = (lane & 16) ? 8 : 0;
  const int bhalf = (lane & 16) ? 8 : 0;

  const float* Abase = X + (size_t)rowBase * K;
  const float* Bbase = W + (size_t)colBase * K;

  v8f zf = {};
  v8f acc[2][2];
#pragma unroll
  for (int i = 0; i < 2; ++i)
#pragma unroll
    for (int j = 0; j < 2; ++j) acc[i][j] = zf;

  auto stageA = [&](int buf, int kk) {
#pragma unroll
    for (int i = 0; i < 4; ++i) {
      int idx = t + i * 128, r = idx >> 3, c4 = (idx & 7) * 4;
#if HAVE_ASYNC
      async_cp16(Abase + (size_t)r * K + kk + c4, &Asf[buf][r][c4]);
#else
      *(float4*)&Asf[buf][r][c4] = *(const float4*)(Abase + (size_t)r * K + kk + c4);
#endif
    }
  };
  auto stageB = [&](int buf, int kk) {
#if HAVE_TDM
    if (w == 0) tdm_load_tile(&Bsf[buf][0][0], Bbase + kk, 64u, 32u, (unsigned)K);
#else
#pragma unroll
    for (int i = 0; i < 4; ++i) {
      int idx = t + i * 128, r = idx >> 3, c4 = (idx & 7) * 4;
#if HAVE_ASYNC
      async_cp16(Bbase + (size_t)r * K + kk + c4, &Bsf[buf][r][c4]);
#else
      *(float4*)&Bsf[buf][r][c4] = *(const float4*)(Bbase + (size_t)r * K + kk + c4);
#endif
    }
#endif
  };
  // async ops per thread per stage (in-order completion => partial waits are safe)
  constexpr int APS = HAVE_TDM ? 4 : 8;

  const int nk = K >> 5;
  stageA(0, 0);
  stageB(0, 0);
  for (int kt = 0; kt < nk; ++kt) {
    const int cur = kt & 1;
    const bool more = (kt + 1 < nk);
    if (more) { stageA(cur ^ 1, (kt + 1) << 5); stageB(cur ^ 1, (kt + 1) << 5); }
#if HAVE_ASYNC
    if (more) wait_async<APS>(); else wait_async<0>();
#endif
#if HAVE_TDM
    if (w == 0) {
      if (more) __builtin_amdgcn_s_wait_tensorcnt(1);
      else      __builtin_amdgcn_s_wait_tensorcnt(0);
    }
#endif
    __syncthreads();

    v16bf af[2], bfv[2];
#pragma unroll
    for (int rg = 0; rg < 2; ++rg) af[rg]  = a_frag_f32(&Asf[cur][wrow + rg * 16 + lrow][0], ahalf);
#pragma unroll
    for (int cg = 0; cg < 2; ++cg) bfv[cg] = b_frag_f32(&Bsf[cur][wcol + cg * 16 + lrow][bhalf]);
#pragma unroll
    for (int rg = 0; rg < 2; ++rg)
#pragma unroll
      for (int cg = 0; cg < 2; ++cg)
        acc[rg][cg] = wmma_bf16(af[rg], bfv[cg], acc[rg][cg]);
    __syncthreads();   // done reading buf[cur] before next iteration refills it
  }

#pragma unroll
  for (int rg = 0; rg < 2; ++rg)
#pragma unroll
    for (int cg = 0; cg < 2; ++cg)
#pragma unroll
      for (int r = 0; r < 8; ++r) {
        int gr = rowBase + wrow + rg * 16 + r + ahalf;
        int gc = colBase + wcol + cg * 16 + lrow;
        float val = acc[rg][cg][r] + bias[gc];
        if (mode == 0) {
          Y[(size_t)gr * N + gc] = val;
        } else {
          int b = gr >> 11, m = gr & (MSEQ - 1);
          int h = gc >> 6,  d = gc & (HEAD - 1);
          Y[(((size_t)b * NUM_HEADS + h) * MSEQ + m) * HEAD + d] = val * scale;
        }
      }
}

// Flash attention: grid (M/128, B*H), 128 threads (4 waves); each wave owns 32 query
// rows, streams 32-key blocks with online softmax. K staged async-to-LDS (fp32),
// V staged transposed as bf16, P transposed through per-wave LDS. Q pre-scaled 1/8.
__global__ __launch_bounds__(128) void flash_attn(const float* __restrict__ Q,
                                                  const float* __restrict__ Kh,
                                                  const float* __restrict__ V,
                                                  const unsigned char* __restrict__ mask,
                                                  float* __restrict__ ctx) {
  __shared__ float  Ksf[32][68];      // 32 keys x 64 d (fp32, +4 pad)
  __shared__ bf16_t Vt[64][40];       // transposed: 64 d x 32 keys
  __shared__ bf16_t Pw[4][32][40];    // per-wave P block (32 q x 32 k)

  const int t = threadIdx.x, lane = t & 31, w = t >> 5;
  const int head = blockIdx.y;                 // b*H + h
  const int b = head >> 4, h = head & 15;
  const size_t hb = (size_t)head * MSEQ * HEAD;
  const int qbase = blockIdx.x * 128 + w * 32;
  const unsigned char* mrow = mask + (size_t)b * MSEQ;
  const int lrow = lane & 15;
  const int ahalf = (lane & 16) ? 8 : 0;
  const int bhalf = (lane & 16) ? 8 : 0;

  // Q fragments (2 row-groups x 2 K-steps of 32), loaded once
  v16bf qf[2][2];
#pragma unroll
  for (int rg = 0; rg < 2; ++rg) {
    const float* qp = Q + hb + (size_t)(qbase + rg * 16 + lrow) * HEAD;
#pragma unroll
    for (int ks = 0; ks < 2; ++ks) {
      int koff = ks * 32 + ahalf;
      v16bf f;
#pragma unroll
      for (int i = 0; i < 8; ++i) { f[i] = to_bf16(qp[koff + i]); f[i + 8] = to_bf16(qp[koff + 16 + i]); }
      qf[rg][ks] = f;
    }
  }

  v8f zf = {};
  v8f oacc[2][4];
  float mrun[2][8], lrun[2][8];
#pragma unroll
  for (int rg = 0; rg < 2; ++rg) {
#pragma unroll
    for (int c = 0; c < 4; ++c) oacc[rg][c] = zf;
#pragma unroll
    for (int r = 0; r < 8; ++r) { mrun[rg][r] = -1e30f; lrun[rg][r] = 0.f; }
  }

  for (int kb = 0; kb < MSEQ; kb += 32) {
#pragma unroll
    for (int i = 0; i < 4; ++i) {
      int idx = t + i * 128, r = idx >> 4, c4 = (idx & 15) * 4;
#if HAVE_ASYNC
      async_cp16(Kh + hb + (size_t)(kb + r) * HEAD + c4, &Ksf[r][c4]);
#else
      *(float4*)&Ksf[r][c4] = *(const float4*)(Kh + hb + (size_t)(kb + r) * HEAD + c4);
#endif
      const float4 v4 = *(const float4*)(V + hb + (size_t)(kb + r) * HEAD + c4);
      Vt[c4 + 0][r] = to_bf16(v4.x); Vt[c4 + 1][r] = to_bf16(v4.y);
      Vt[c4 + 2][r] = to_bf16(v4.z); Vt[c4 + 3][r] = to_bf16(v4.w);
    }
#if HAVE_ASYNC
    wait_async<0>();
#endif
    __syncthreads();

    // S = Q @ K^T
    v8f sacc[2][2];
#pragma unroll
    for (int rg = 0; rg < 2; ++rg)
#pragma unroll
      for (int cg = 0; cg < 2; ++cg) sacc[rg][cg] = zf;
#pragma unroll
    for (int ks = 0; ks < 2; ++ks) {
      v16bf bk[2];
#pragma unroll
      for (int cg = 0; cg < 2; ++cg)
        bk[cg] = b_frag_f32(&Ksf[cg * 16 + lrow][ks * 32 + bhalf]);
#pragma unroll
      for (int rg = 0; rg < 2; ++rg)
#pragma unroll
        for (int cg = 0; cg < 2; ++cg)
          sacc[rg][cg] = wmma_bf16(qf[rg][ks], bk[cg], sacc[rg][cg]);
    }

    const float M0 = mrow[kb + lrow]      ? 0.f : -1e30f;
    const float M1 = mrow[kb + 16 + lrow] ? 0.f : -1e30f;

    // Online softmax (row = 16 lanes of one half at fixed frag index r)
#pragma unroll
    for (int rg = 0; rg < 2; ++rg)
#pragma unroll
      for (int r = 0; r < 8; ++r) {
        float s0 = sacc[rg][0][r] + M0;
        float s1 = sacc[rg][1][r] + M1;
        float mx = fmaxf(s0, s1);
#pragma unroll
        for (int d = 1; d < 16; d <<= 1) mx = fmaxf(mx, __shfl_xor(mx, d, 32));
        float mold = mrun[rg][r];
        float mnew = fmaxf(mold, mx);
        float alpha = __expf(mold - mnew);
        float p0 = __expf(s0 - mnew);
        float p1 = __expf(s1 - mnew);
        float ls = p0 + p1;
#pragma unroll
        for (int d = 1; d < 16; d <<= 1) ls += __shfl_xor(ls, d, 32);
        mrun[rg][r] = mnew;
        lrun[rg][r] = lrun[rg][r] * alpha + ls;
#pragma unroll
        for (int c = 0; c < 4; ++c) oacc[rg][c][r] *= alpha;
        int prow = rg * 16 + r + ahalf;
        Pw[w][prow][lrow]      = to_bf16(p0);
        Pw[w][prow][16 + lrow] = to_bf16(p1);
      }

    // O += P @ V  (per-wave LDS transpose; same-wave DS ops are in-order)
    v16bf ap[2];
#pragma unroll
    for (int rg = 0; rg < 2; ++rg)
      ap[rg] = a_frag(&Pw[w][rg * 16 + lrow][0], ahalf);
#pragma unroll
    for (int cg = 0; cg < 4; ++cg) {
      v16bf bv = b_frag(&Vt[cg * 16 + lrow][bhalf]);
#pragma unroll
      for (int rg = 0; rg < 2; ++rg)
        oacc[rg][cg] = wmma_bf16(ap[rg], bv, oacc[rg][cg]);
    }
    __syncthreads();
  }

#pragma unroll
  for (int rg = 0; rg < 2; ++rg)
#pragma unroll
    for (int cg = 0; cg < 4; ++cg)
#pragma unroll
      for (int r = 0; r < 8; ++r) {
        int m = qbase + rg * 16 + r + ahalf;
        int d = cg * 16 + lrow;
        float l = lrun[rg][r];
        float val = oacc[rg][cg][r] / (l > 0.f ? l : 1.f);
        ctx[((size_t)b * MSEQ + m) * DMODEL + h * HEAD + d] = val;
      }
}

extern "C" void kernel_launch(void* const* d_in, const int* in_sizes, int n_in,
                              void* d_out, int out_size, void* d_ws, size_t ws_size,
                              hipStream_t stream) {
  (void)in_sizes; (void)n_in; (void)out_size; (void)ws_size;
  const float* k  = (const float*)d_in[0];
  const float* v  = (const float*)d_in[1];
  const float* q  = (const float*)d_in[2];
  const unsigned char* mask = (const unsigned char*)d_in[3];
  const float* Wk = (const float*)d_in[4];
  const float* bk = (const float*)d_in[5];
  const float* Wv = (const float*)d_in[6];
  const float* bv = (const float*)d_in[7];
  const float* Wq = (const float*)d_in[8];
  const float* bq = (const float*)d_in[9];
  const float* Wo = (const float*)d_in[10];
  const float* bo = (const float*)d_in[11];
  float* out = (float*)d_out;

  const size_t elems = (size_t)BATCH * MSEQ * DMODEL;  // 4M floats = 16MB
  float* Kh  = (float*)d_ws;
  float* Vh  = Kh + elems;
  float* Qh  = Vh + elems;
  float* ctx = Qh + elems;

  dim3 blk(128);
  dim3 g1(BATCH * MSEQ / 64, DMODEL / 64);   // 64 x 16 tiles
  proj_gemm<<<g1, blk, 0, stream>>>(k, Wk, bk, Kh, DMODEL, DMODEL, 1, 1.0f);
  proj_gemm<<<g1, blk, 0, stream>>>(v, Wv, bv, Vh, DMODEL, DMODEL, 1, 1.0f);
  proj_gemm<<<g1, blk, 0, stream>>>(q, Wq, bq, Qh, DMODEL, DMODEL, 1, 0.125f);

  dim3 g2(MSEQ / 128, BATCH * NUM_HEADS);    // 16 x 32
  flash_attn<<<g2, blk, 0, stream>>>(Qh, Kh, Vh, mask, ctx);

  proj_gemm<<<g1, blk, 0, stream>>>(ctx, Wo, bo, out, DMODEL, DMODEL, 0, 1.0f);
}
